// MambaBlock_1073741824651
// MI455X (gfx1250) — matrix operand order repaired and checked by reading, not compiled, not run on hardware
//
#include <hip/hip_runtime.h>
#include <hip/hip_bf16.h>

// ---------- types ----------
typedef __bf16 bf16_t;
typedef __attribute__((ext_vector_type(16))) __bf16 v16bf;
typedef __attribute__((ext_vector_type(8)))  float  v8f;

#define LSEQ 1024
#define DMODEL 1024
#define DINNER 2048
#define NSTATE 16
#define DTRANK 64
#define KCONV 4

// register blocking: each wave computes (BM*16) x (BN*16) of C
#define BM 4
#define BN 2

// ---------- helpers ----------
__device__ __forceinline__ unsigned short f2bf_u(float f) {
    unsigned u = __float_as_uint(f);
    u += 0x7FFFu + ((u >> 16) & 1u);   // round-to-nearest-even
    return (unsigned short)(u >> 16);
}
__device__ __forceinline__ bf16_t f2bf(float f) {
    union { unsigned short s; bf16_t b; } x; x.s = f2bf_u(f); return x.b;
}
__device__ __forceinline__ float silu_f(float x) {
    return x / (1.0f + __expf(-x));
}

// ---------- fp32 [K x N] -> bf16 [N x K] (transpose + convert), LDS tiled ----------
__global__ void cvt_transpose_kernel(const float* __restrict__ in, bf16_t* __restrict__ out,
                                     int K, int N) {
    __shared__ unsigned short tile[32][33];
    int nb = blockIdx.x * 32, kb = blockIdx.y * 32;
    int tx = threadIdx.x, ty = threadIdx.y;        // block 32x8
    #pragma unroll
    for (int i = ty; i < 32; i += 8) {
        int k = kb + i, n = nb + tx;
        unsigned short h = 0;
        if (k < K && n < N) h = f2bf_u(in[(size_t)k * N + n]);
        tile[i][tx] = h;
    }
    __syncthreads();
    #pragma unroll
    for (int i = ty; i < 32; i += 8) {
        int n = nb + i, k = kb + tx;
        if (n < N && k < K) {
            union { unsigned short s; bf16_t b; } x; x.s = tile[tx][i];
            out[(size_t)n * K + k] = x.b;
        }
    }
}

// ---------- fp32 [rows x instride] (take first `cols`) -> bf16 [rows x cols] ----------
__global__ void cvt_bf16_kernel(const float* __restrict__ in, bf16_t* __restrict__ out,
                                int rows, int cols, int instride) {
    int idx = blockIdx.x * 256 + threadIdx.x;
    if (idx >= rows * cols) return;
    int r = idx / cols, c = idx - r * cols;
    out[idx] = f2bf(in[(size_t)r * instride + c]);
}

// ---------- bf16 WMMA GEMM: C[MxN] = A[MxK] * Bt[NxK]^T  (fp32 accumulate) ----------
// Each wave computes a 64x32 block of C (4x2 grid of 16x16 WMMA tiles).
// mode 0: plain store.  mode 1: C = softplus(acc + bias[n])
__global__ void gemm_bf16_wmma_kernel(const bf16_t* __restrict__ A,
                                      const bf16_t* __restrict__ Bt,
                                      float* __restrict__ C,
                                      const float* __restrict__ bias,
                                      int M, int N, int K, int mode) {
    int wave = (blockIdx.x * blockDim.x + threadIdx.x) >> 5;
    int lane = threadIdx.x & 31;
    int wTilesN = N >> 5;                       // 32-wide super-tiles
    int wTilesTot = (M >> 6) * wTilesN;         // 64-tall super-tiles
    if (wave >= wTilesTot) return;
    int tm = (wave / wTilesN) << 6;
    int tn = (wave % wTilesN) << 5;
    int half = lane >> 4;       // 0: lanes 0-15, 1: lanes 16-31
    int mr   = lane & 15;       // row of A fragment / row of Bt (output column)

    const bf16_t* Arow[BM];
    #pragma unroll
    for (int i = 0; i < BM; ++i) Arow[i] = A + (size_t)(tm + i * 16 + mr) * K;
    const bf16_t* Brow[BN];
    #pragma unroll
    for (int j = 0; j < BN; ++j) Brow[j] = Bt + (size_t)(tn + j * 16 + mr) * K;

    v8f acc[BM][BN];
    #pragma unroll
    for (int i = 0; i < BM; ++i)
        #pragma unroll
        for (int j = 0; j < BN; ++j) acc[i][j] = (v8f){};

    for (int k0 = 0; k0 < K; k0 += 32) {
        union F { unsigned u[8]; v16bf v; };
        F a[BM];
        #pragma unroll
        for (int i = 0; i < BM; ++i) {
            #pragma unroll
            for (int v = 0; v < 8; ++v) {
                // 16-bit A 16x32 layout: dword v -> K = (v>>2)*16 + half*8 + (v&3)*2
                int ka = ((v >> 2) << 4) + (half << 3) + ((v & 3) << 1);
                a[i].u[v] = *(const unsigned*)(Arow[i] + k0 + ka);
            }
        }
        #pragma unroll
        for (int j = 0; j < BN; ++j) {
            F b;
            #pragma unroll
            for (int v = 0; v < 8; ++v) {
                // 16-bit B 32x16 layout: dword v -> K = (v>>2)*16 + half*8 + (v&3)*2 (N = lane&15)
                int kb = ((v >> 2) << 4) + (half << 3) + ((v & 3) << 1);
                b.u[v] = *(const unsigned*)(Brow[j] + k0 + kb);
            }
            #pragma unroll
            for (int i = 0; i < BM; ++i) {
                acc[i][j] = __builtin_amdgcn_wmma_f32_16x16x32_bf16(
                                false, a[i].v, false, b.v, (short)0, acc[i][j], false, false);
            }
        }
    }

    #pragma unroll
    for (int j = 0; j < BN; ++j) {
        int n = tn + j * 16 + mr;
        float bval = (mode == 1) ? bias[n] : 0.0f;
        #pragma unroll
        for (int i = 0; i < BM; ++i) {
            #pragma unroll
            for (int r = 0; r < 8; ++r) {
                int m = tm + i * 16 + r + (half << 3);  // C/D layout: VGPR r -> M = r + half*8
                float val = acc[i][j][r];
                if (mode == 1) {
                    val += bval;
                    val = (val > 20.0f) ? val : log1pf(__expf(val));   // softplus
                }
                C[(size_t)m * N + n] = val;
            }
        }
    }
}

// ---------- causal depthwise conv (K=4) + SiLU; emits fp32 and bf16 ----------
__global__ void conv_silu_kernel(const float* __restrict__ xw,   // [L x 4096], x_in = cols 0..2047
                                 const float* __restrict__ cw,   // [2048 x 4]
                                 const float* __restrict__ cb,   // [2048]
                                 float* __restrict__ xc_f,       // [L x 2048]
                                 bf16_t* __restrict__ xc_b) {    // [L x 2048]
    int idx = blockIdx.x * 256 + threadIdx.x;      // L*DINNER threads
    int c = idx & (DINNER - 1);
    int l = idx >> 11;
    float s = cb[c];
    #pragma unroll
    for (int k = 0; k < KCONV; ++k) {
        int li = l + k - (KCONV - 1);
        if (li >= 0) s = fmaf(xw[(size_t)li * (2 * DINNER) + c], cw[c * KCONV + k], s);
    }
    float o = silu_f(s);
    xc_f[idx] = o;
    xc_b[idx] = f2bf(o);
}

// ---------- selective scan: lane = (d, n); 2 channels per wave32; shfl_xor reduce ----------
__global__ void mamba_scan_kernel(const float* __restrict__ delta,  // [L x DI]
                                  const float* __restrict__ xconv,  // [L x DI] (u)
                                  const float* __restrict__ xdbl,   // [L x 96] (B at 64, C at 80)
                                  const float* __restrict__ xw,     // [L x 4096] (res at col 2048)
                                  const float* __restrict__ A_log,  // [DI x 16]
                                  const float* __restrict__ Dp,     // [DI]
                                  bf16_t* __restrict__ ygated) {    // [L x DI]
    int g = blockIdx.x * blockDim.x + threadIdx.x;   // 0 .. DI*NSTATE-1
    int n = g & (NSTATE - 1);
    int d = g >> 4;
    float Aval = -__expf(A_log[d * NSTATE + n]);
    float Dd = Dp[d];
    float h = 0.0f;
    for (int l = 0; l < LSEQ; ++l) {
        float dl = delta[(size_t)l * DINNER + d];
        float u  = xconv[(size_t)l * DINNER + d];
        float Bn = xdbl[(size_t)l * 96 + DTRANK + n];
        float Cn = xdbl[(size_t)l * 96 + DTRANK + NSTATE + n];
        float da = __expf(dl * Aval);
        h = fmaf(da, h, dl * Bn * u);
        float p = h * Cn;
        p += __shfl_xor(p, 1, 32);
        p += __shfl_xor(p, 2, 32);
        p += __shfl_xor(p, 4, 32);
        p += __shfl_xor(p, 8, 32);   // 16-lane tree sum, stays within each half-wave
        if (n == 0) {
            float y = p + u * Dd;
            float r = xw[(size_t)l * (2 * DINNER) + DINNER + d];
            ygated[(size_t)l * DINNER + d] = f2bf(y * silu_f(r));
        }
    }
}

// ---------- launch ----------
extern "C" void kernel_launch(void* const* d_in, const int* in_sizes, int n_in,
                              void* d_out, int out_size, void* d_ws, size_t ws_size,
                              hipStream_t stream) {
    const float* x      = (const float*)d_in[0];   // [L, DM]
    const float* W_in   = (const float*)d_in[1];   // [DM, 2*DI]
    const float* conv_w = (const float*)d_in[2];   // [DI, 4]
    const float* conv_b = (const float*)d_in[3];   // [DI]
    const float* W_x    = (const float*)d_in[4];   // [DI, 96]
    const float* W_dt   = (const float*)d_in[5];   // [64, DI]
    const float* b_dt   = (const float*)d_in[6];   // [DI]
    const float* A_log  = (const float*)d_in[7];   // [DI, 16]
    const float* Dvec   = (const float*)d_in[8];   // [DI]
    const float* W_out  = (const float*)d_in[9];   // [DI, DM]
    float* out = (float*)d_out;                    // [L, DM]

    char* w = (char*)d_ws;
    bf16_t* WinT  = (bf16_t*)(w + 0);            // [4096 x 1024]  8.0 MB
    bf16_t* WxT   = (bf16_t*)(w + 8388608);      // [96 x 2048]    0.375 MB
    bf16_t* WdtT  = (bf16_t*)(w + 8781824);      // [2048 x 64]    0.25 MB
    bf16_t* WoutT = (bf16_t*)(w + 9043968);      // [1024 x 2048]  4 MB
    bf16_t* xb    = (bf16_t*)(w + 13238272);     // [1024 x 1024]  2 MB
    float*  xw    = (float*) (w + 15335424);     // [1024 x 4096] 16 MB
    float*  xc_f  = (float*) (w + 32112640);     // [1024 x 2048]  8 MB
    bf16_t* xc_b  = (bf16_t*)(w + 40501248);     // [1024 x 2048]  4 MB
    float*  xdbl  = (float*) (w + 44695552);     // [1024 x 96]    0.375 MB
    bf16_t* dtb   = (bf16_t*)(w + 45088768);     // [1024 x 64]    0.125 MB
    float*  delta = (float*) (w + 45219840);     // [1024 x 2048]  8 MB
    bf16_t* yg    = (bf16_t*)(w + 53608448);     // [1024 x 2048]  4 MB

    dim3 tb32x8(32, 8);
    // Weight convert + transpose: in [K x N] -> out [N x K]
    cvt_transpose_kernel<<<dim3(4096/32, 1024/32), tb32x8, 0, stream>>>(W_in,  WinT,  1024, 4096);
    cvt_transpose_kernel<<<dim3(  96/32, 2048/32), tb32x8, 0, stream>>>(W_x,   WxT,   2048,   96);
    cvt_transpose_kernel<<<dim3(2048/32,   64/32), tb32x8, 0, stream>>>(W_dt,  WdtT,    64, 2048);
    cvt_transpose_kernel<<<dim3(1024/32, 2048/32), tb32x8, 0, stream>>>(W_out, WoutT, 2048, 1024);
    // x -> bf16
    cvt_bf16_kernel<<<(LSEQ*DMODEL)/256, 256, 0, stream>>>(x, xb, LSEQ, DMODEL, DMODEL);

    // GEMM1: xw = x @ W_in   [1024 x 4096], K=1024  -> (16 x 128) super-tiles = 2048 waves
    gemm_bf16_wmma_kernel<<<2048/8, 256, 0, stream>>>(xb, WinT, xw, nullptr,
                                                      LSEQ, 2*DINNER, DMODEL, 0);
    // conv + silu
    conv_silu_kernel<<<(LSEQ*DINNER)/256, 256, 0, stream>>>(xw, conv_w, conv_b, xc_f, xc_b);

    // GEMM2: x_dbl = x_conv @ W_x  [1024 x 96], K=2048 -> (16 x 3) = 48 waves
    gemm_bf16_wmma_kernel<<<48/8, 256, 0, stream>>>(xc_b, WxT, xdbl, nullptr,
                                                    LSEQ, DTRANK + 2*NSTATE, DINNER, 0);
    // dt slice -> bf16
    cvt_bf16_kernel<<<(LSEQ*DTRANK)/256, 256, 0, stream>>>(xdbl, dtb, LSEQ, DTRANK, 96);

    // GEMM3: delta = softplus(dt @ W_dt + b_dt)  [1024 x 2048], K=64 -> (16 x 64) = 1024 waves
    gemm_bf16_wmma_kernel<<<1024/8, 256, 0, stream>>>(dtb, WdtT, delta, b_dt,
                                                      LSEQ, DINNER, DTRANK, 1);
    // selective scan + gating, bf16 output
    mamba_scan_kernel<<<(DINNER*NSTATE)/256, 256, 0, stream>>>(delta, xc_f, xdbl, xw,
                                                               A_log, Dvec, yg);
    // GEMM4: out = ygated @ W_out  [1024 x 1024], K=2048 -> (16 x 32) = 512 waves
    gemm_bf16_wmma_kernel<<<512/8, 256, 0, stream>>>(yg, WoutT, out, nullptr,
                                                     LSEQ, DMODEL, DINNER, 0);
}
